// Attention_41884521070695
// MI455X (gfx1250) — matrix operand order repaired and checked by reading, not compiled, not run on hardware
//
#include <hip/hip_runtime.h>
#include <hip/hip_bf16.h>
#include <cstdint>

// ---------------------------------------------------------------------------
// Fused MHA forward for B=16, S=1024, D=768, H=12, HD=64 on gfx1250 (wave32).
// - All matmuls: v_wmma_f32_16x16x32_f16 (f16 in, f32 accumulate).
// - All WMMA fragments are contiguous per-lane 32B LDS reads (ds_load_b128).
// - Attention Q/K tiles staged by the Tensor Data Mover (tensor_load_to_lds,
//   TENSORcnt) with hardware LDS padding; manual fallback if builtin absent.
// ---------------------------------------------------------------------------

typedef _Float16 half_t;
typedef __attribute__((ext_vector_type(2)))  _Float16 v2h;
typedef __attribute__((ext_vector_type(16))) _Float16 v16h;
typedef __attribute__((ext_vector_type(8)))  float    v8f;

#define B_  16
#define S_  1024
#define D_  768
#define H_  12
#define HD_ 64
#define BS_ (B_ * S_)

#if defined(__has_builtin)
#if __has_builtin(__builtin_amdgcn_tensor_load_to_lds)
#define HAVE_TDM 1
#endif
#endif
#if __has_include(<hip/amd_detail/amd_gfx1250_TDM.h>)
#define TDM_6ARG 1   // therock-10.0 headers -> clang-23 6-arg builtin
#endif

__device__ __forceinline__ v8f wmma_f32_16x16x32_f16(v16h a, v16h b, v8f c) {
  return __builtin_amdgcn_wmma_f32_16x16x32_f16(
      /*neg_a=*/false, a, /*neg_b=*/false, b,
      /*c_mod=*/(short)0, c, /*reuse_a=*/false, /*reuse_b=*/false);
}

// ---- fragment loaders (wave32; contiguous k order, same for A and B) -------------------------
// A fragment 16x32 (MxK), row-major source. lane m = lane&15;
// k(j) = 16*(lane>>4) + j  -> one contiguous 16-half chunk.
__device__ __forceinline__ v16h lds_load_A(const half_t* s, int row0, int kcol0, int ld) {
  const int lane = threadIdx.x & 31;
  const half_t* p = s + (size_t)(row0 + (lane & 15)) * ld + kcol0 + ((lane >> 4) << 4);
  v16h a;
#pragma unroll
  for (int j = 0; j < 8; ++j)  a[j] = p[j];
#pragma unroll
  for (int j = 0; j < 8; ++j)  a[8 + j] = p[8 + j];
  return a;
}

// B fragment 32x16 (KxN) where the tile is stored N-major: s[n][k] (ld = k pitch).
// lane n = lane&15, k(j) = 16*(lane>>4) + j -> one contiguous 16-half chunk.
__device__ __forceinline__ v16h lds_load_BT(const half_t* s, int nrow0, int kcol0, int ld) {
  const int lane = threadIdx.x & 31;
  const half_t* p = s + (size_t)(nrow0 + (lane & 15)) * ld + kcol0 + ((lane >> 4) << 4);
  v16h b;
#pragma unroll
  for (int j = 0; j < 16; ++j) b[j] = p[j];
  return b;
}

// ---- Tensor Data Mover: 1-D f16 span global->LDS with HW padding -----------
// Pads 4 DWORDs (8 halves) after every 32 DWORDs (64 halves): LDS pitch 72.
#if defined(HAVE_TDM)
typedef __attribute__((ext_vector_type(4))) unsigned int tdm_v4u;
typedef __attribute__((ext_vector_type(8))) int          tdm_v8i;
typedef __attribute__((ext_vector_type(4))) int          tdm_v4i;

__device__ __forceinline__ void tdm_load_pad64(void* lds_dst, const void* gsrc,
                                               unsigned int nhalves) {
  const unsigned long long ga = (unsigned long long)(uintptr_t)gsrc;
  tdm_v4u g0;
  g0[0] = 1u;                                          // count=1, user descriptor
  g0[1] = (unsigned int)(uintptr_t)lds_dst;            // lds_addr = flat[31:0]
  g0[2] = (unsigned int)(ga & 0xFFFFFFFFu);            // global_addr[31:0]
  g0[3] = (unsigned int)((ga >> 32) & 0x01FFFFFFu)     // global_addr[56:32]
        | (2u << 30);                                  // type = 2 (image)
  tdm_v8i g1;
  g1[0] = (int)((1u << 16)        // data_size = 1 -> 2 bytes
              | (1u << 20)        // pad_enable
              | (4u << 22)        // pad_interval: 32 DWORDs (64 halves)
              | (3u << 25));      // pad_amount: 4 DWORDs (8 halves)
  g1[1] = (int)((0x7FFFFFFFu & 0xFFFFu) << 16);        // tensor_dim0[15:0]
  g1[2] = (int)(0x7FFFFFFFu >> 16);                    // tensor_dim0[31:16]
  g1[3] = (int)((nhalves & 0xFFFFu) << 16);            // tile_dim0
  g1[4] = 0;                                           // tile_dim1/2 = 0 (1-D)
  g1[5] = 0; g1[6] = 0; g1[7] = 0;
  tdm_v4i z4 = {};
#if defined(TDM_6ARG)
  tdm_v8i z8 = {};
  __builtin_amdgcn_tensor_load_to_lds(g0, g1, z4, z4, z8, 0);
#else
  __builtin_amdgcn_tensor_load_to_lds(g0, g1, z4, z4, 0);
#endif
}
#endif  // HAVE_TDM

// ---------------------------------------------------------------------------
// Kernel 1: QKV projection.  X[BS,768] (f32) x W[768,768] (f32) + bias -> f16
// in [B,H,S,HD] layout, scaled (q gets 1/sqrt(HD)).
// Block: 256 thr (8 waves); tile 128x64; wave tile 32x32 (4 WMMA / k-step).
// (fp32 sources must be converted, so these tiles are staged by VALU.)
// ---------------------------------------------------------------------------
#define LDA  40   // 32 k + 8 pad (halves)
#define LDBT 40   // Bt[n][k] pitch

__global__ __launch_bounds__(256) void qkv_proj_kernel(
    const float* __restrict__ X, const float* __restrict__ W,
    const float* __restrict__ bias, half_t* __restrict__ out, float scale) {
  __shared__ __align__(16) half_t As[128 * LDA];
  __shared__ __align__(16) half_t Bt[64 * LDBT];    // transposed W tile: [n][k]
  const int t = threadIdx.x;
  const int wave = t >> 5, lane = t & 31;
  const int M0 = blockIdx.x * 128;
  const int N0 = blockIdx.y * 64;
  const int wm = (wave >> 1) * 32;   // 0/32/64/96
  const int wn = (wave & 1) * 32;    // 0/32
  v8f acc00 = {}, acc01 = {}, acc10 = {}, acc11 = {};

  for (int k0 = 0; k0 < D_; k0 += 32) {
    __syncthreads();
#pragma unroll
    for (int i = 0; i < 8; ++i) {              // A tile 128x32, float2 -> packed b32
      int idx = t + 256 * i;                   // 2048 float2
      int r = idx >> 4, cp = idx & 15;
      const float* g = &X[(size_t)(M0 + r) * D_ + k0 + cp * 2];
      v2h h; h[0] = (half_t)g[0]; h[1] = (half_t)g[1];
      *(v2h*)&As[r * LDA + cp * 2] = h;
    }
#pragma unroll
    for (int i = 0; i < 4; ++i) {              // B tile 32x64 -> transposed store
      int idx = t + 256 * i;                   // 1024 float2
      int r = idx >> 5, cp = idx & 31;         // r = k row, cp = n pair
      const float* g = &W[(size_t)(k0 + r) * D_ + N0 + cp * 2];
      Bt[(cp * 2 + 0) * LDBT + r] = (half_t)g[0];
      Bt[(cp * 2 + 1) * LDBT + r] = (half_t)g[1];
    }
    __syncthreads();
    v16h a0 = lds_load_A(As, wm, 0, LDA);
    v16h a1 = lds_load_A(As, wm + 16, 0, LDA);
    v16h b0 = lds_load_BT(Bt, wn, 0, LDBT);
    v16h b1 = lds_load_BT(Bt, wn + 16, 0, LDBT);
    acc00 = wmma_f32_16x16x32_f16(a0, b0, acc00);
    acc01 = wmma_f32_16x16x32_f16(a0, b1, acc01);
    acc10 = wmma_f32_16x16x32_f16(a1, b0, acc10);
    acc11 = wmma_f32_16x16x32_f16(a1, b1, acc11);
  }

  const int n = lane & 15;
  const int mh = (lane >> 4) << 3;
#pragma unroll
  for (int mi = 0; mi < 2; ++mi) {
#pragma unroll
    for (int r = 0; r < 8; ++r) {
      int m = M0 + wm + mi * 16 + mh + r;
      int b = m >> 10;                 // / S_
      int s = m & (S_ - 1);
#pragma unroll
      for (int ni = 0; ni < 2; ++ni) {
        const v8f& acc = mi ? (ni ? acc11 : acc10) : (ni ? acc01 : acc00);
        int col = N0 + wn + ni * 16 + n;
        float v = (acc[r] + bias[col]) * scale;
        int h = col >> 6, e = col & 63;
        out[((size_t)(b * H_ + h) * S_ + s) * HD_ + e] = (half_t)v;
      }
    }
  }
}

// ---------------------------------------------------------------------------
// Kernel 2: flash attention.  Q,K,V f16 in [B,H,S,64] (Q pre-scaled).
// Block: 256 thr; 128 q-rows per block (16 per wave); KV tiles of 32.
// Q/K tiles: TDM (tensor_load_to_lds + s_wait_tensorcnt), HW-padded to pitch 72.
// V tile: manual transposed staging ([e][kv]; V fragments contiguous).
// ---------------------------------------------------------------------------
#define LQ  72    // Q/K tile pitch (64 + 8) == TDM pad pattern
#define LPV 40    // Vt / P pitch (32 + 8)

__global__ __launch_bounds__(256) void attn_kernel(
    const half_t* __restrict__ Q, const half_t* __restrict__ K,
    const half_t* __restrict__ V, half_t* __restrict__ Oh) {
  __shared__ __align__(16) half_t Qs[128 * LQ];
  __shared__ __align__(16) half_t Ks[32 * LQ];
  __shared__ __align__(16) half_t Vt[64 * LPV];      // transposed V tile: [e][kv]
  __shared__ __align__(16) half_t Ps[8 * 16 * LPV];  // per-wave P scratch

  const int t = threadIdx.x, wave = t >> 5, lane = t & 31;
  const int bh = blockIdx.x;                 // b*H + h
  const int b = bh / H_, h = bh % H_;
  const int q0 = blockIdx.y * 128;
  const size_t base = (size_t)bh * S_ * HD_;

#if defined(HAVE_TDM)
  if (t < 32) {  // wave 0 kicks off the async Q tile copy (128x64, pitch 72)
    tdm_load_pad64(Qs, Q + base + (size_t)q0 * HD_, 128 * HD_);
  }
#else
  {  // manual Q tile staging (u32 pairs)
    const uint32_t* src = (const uint32_t*)(Q + base + (size_t)q0 * HD_);
#pragma unroll
    for (int i = 0; i < 16; ++i) {
      int idx = t + 256 * i;                 // 4096 u32
      int r = idx >> 5, cp = idx & 31;
      *(uint32_t*)&Qs[r * LQ + cp * 2] = src[idx];
    }
  }
#endif

  float rowmax[8], rowsum[8];
  v8f Oacc[4];
#pragma unroll
  for (int r = 0; r < 8; ++r) { rowmax[r] = -1e30f; rowsum[r] = 0.f; }
#pragma unroll
  for (int j = 0; j < 4; ++j) Oacc[j] = (v8f){};

  half_t* Pw = Ps + wave * 16 * LPV;

  for (int kv0 = 0; kv0 < S_; kv0 += 32) {
    __syncthreads();   // everyone done reading Ks/Vt from previous tile
#if defined(HAVE_TDM)
    if (t < 32) {      // async K tile copy (32x64, pitch 72) while others stage V
      tdm_load_pad64(Ks, K + base + (size_t)kv0 * HD_, 32 * HD_);
    }
#endif
    {  // stage V (transposed) 32x64 tile
      const uint32_t* vs = (const uint32_t*)(V + base + (size_t)kv0 * HD_);
#if !defined(HAVE_TDM)
      const uint32_t* ks = (const uint32_t*)(K + base + (size_t)kv0 * HD_);
#endif
#pragma unroll
      for (int i = 0; i < 4; ++i) {
        int idx = t + 256 * i;               // 1024 u32 each
        int r = idx >> 5, cp = idx & 31;     // r = kv row, cp = e pair
#if !defined(HAVE_TDM)
        *(uint32_t*)&Ks[r * LQ + cp * 2] = ks[idx];
#endif
        union { uint32_t u; half_t h[2]; } cv;
        cv.u = vs[idx];
        Vt[(cp * 2 + 0) * LPV + r] = cv.h[0];
        Vt[(cp * 2 + 1) * LPV + r] = cv.h[1];
      }
    }
    // prefetch next K/V tiles into cache while we compute on this one
    if (kv0 + 32 < S_) {
      const char* nk = (const char*)(K + base + (size_t)(kv0 + 32) * HD_);
      const char* nv = (const char*)(V + base + (size_t)(kv0 + 32) * HD_);
      __builtin_prefetch(nk + t * 16, 0, 3);
      __builtin_prefetch(nv + t * 16, 0, 3);
    }
#if defined(HAVE_TDM)
    if (t < 32) __builtin_amdgcn_s_wait_tensorcnt(0);  // Q (1st iter) + K done
#endif
    __syncthreads();

    // scores S = Q(16x64) * K^T(64x32) -> two 16x16 frags
    v8f s0 = (v8f){}, s1 = (v8f){};
#pragma unroll
    for (int kk = 0; kk < HD_; kk += 32) {
      v16h a  = lds_load_A(Qs, wave * 16, kk, LQ);
      v16h b0 = lds_load_BT(Ks, 0,  kk, LQ);
      v16h b1 = lds_load_BT(Ks, 16, kk, LQ);
      s0 = wmma_f32_16x16x32_f16(a, b0, s0);
      s1 = wmma_f32_16x16x32_f16(a, b1, s1);
    }

    // online softmax (row m = r + 8*(lane>=16); 16 n-values per half-wave)
#pragma unroll
    for (int r = 0; r < 8; ++r) {
      float v0 = s0[r], v1 = s1[r];
      float mx = fmaxf(v0, v1);
#pragma unroll
      for (int msk = 1; msk < 16; msk <<= 1) mx = fmaxf(mx, __shfl_xor(mx, msk, 32));
      float nm = fmaxf(rowmax[r], mx);
      float alpha = __expf(rowmax[r] - nm);
      rowmax[r] = nm;
      float p0 = __expf(v0 - nm), p1 = __expf(v1 - nm);
      float ps = p0 + p1;
#pragma unroll
      for (int msk = 1; msk < 16; msk <<= 1) ps += __shfl_xor(ps, msk, 32);
      rowsum[r] = rowsum[r] * alpha + ps;
#pragma unroll
      for (int j = 0; j < 4; ++j) Oacc[j][r] *= alpha;
      int mrow = r + ((lane >> 4) << 3);     // C-layout -> row-major P in LDS
      Pw[mrow * LPV + (lane & 15)]      = (half_t)p0;
      Pw[mrow * LPV + 16 + (lane & 15)] = (half_t)p1;
    }

    // O += P(16x32) * V(32x64)   (per-wave scratch; LDS in-order per wave)
    v16h pa = lds_load_A(Pw, 0, 0, LPV);
#pragma unroll
    for (int j = 0; j < 4; ++j) {
      v16h vb = lds_load_BT(Vt, j * 16, 0, LPV);
      Oacc[j] = wmma_f32_16x16x32_f16(pa, vb, Oacc[j]);
    }
  }

  // normalize and store f16 to [B,S,H*HD]
  const int n = lane & 15, mh = (lane >> 4) << 3;
#pragma unroll
  for (int r = 0; r < 8; ++r) {
    float inv = 1.0f / rowsum[r];
    int sq = q0 + wave * 16 + mh + r;
    size_t orow = ((size_t)b * S_ + sq) * D_ + h * HD_;
#pragma unroll
    for (int j = 0; j < 4; ++j)
      Oh[orow + j * 16 + n] = (half_t)(Oacc[j][r] * inv);
  }
}

// ---------------------------------------------------------------------------
// Kernel 3: output projection.  Oh[BS,768] (f16) x Wo[768,768] (f32) + bo -> f32
// Same 128x64 block / 32x32 wave tiling as kernel 1.
// ---------------------------------------------------------------------------
__global__ __launch_bounds__(256) void out_proj_kernel(
    const half_t* __restrict__ Oh, const float* __restrict__ Wo,
    const float* __restrict__ bo, float* __restrict__ Y) {
  __shared__ __align__(16) half_t As[128 * LDA];
  __shared__ __align__(16) half_t Bt[64 * LDBT];
  const int t = threadIdx.x;
  const int wave = t >> 5, lane = t & 31;
  const int M0 = blockIdx.x * 128;
  const int N0 = blockIdx.y * 64;
  const int wm = (wave >> 1) * 32;
  const int wn = (wave & 1) * 32;
  v8f acc00 = {}, acc01 = {}, acc10 = {}, acc11 = {};

  for (int k0 = 0; k0 < D_; k0 += 32) {
    __syncthreads();
    {
      const uint32_t* src = (const uint32_t*)Oh;
#pragma unroll
      for (int i = 0; i < 8; ++i) {            // A tile 128x32 halves (u32 copies)
        int idx = t + 256 * i;                 // 2048 u32
        int r = idx >> 4, cp = idx & 15;
        *(uint32_t*)&As[r * LDA + cp * 2] =
            src[(size_t)(M0 + r) * (D_ / 2) + (k0 >> 1) + cp];
      }
    }
#pragma unroll
    for (int i = 0; i < 4; ++i) {              // B tile 32x64 -> transposed store
      int idx = t + 256 * i;
      int r = idx >> 5, cp = idx & 31;
      const float* g = &Wo[(size_t)(k0 + r) * D_ + N0 + cp * 2];
      Bt[(cp * 2 + 0) * LDBT + r] = (half_t)g[0];
      Bt[(cp * 2 + 1) * LDBT + r] = (half_t)g[1];
    }
    __syncthreads();
    v16h a0 = lds_load_A(As, wm, 0, LDA);
    v16h a1 = lds_load_A(As, wm + 16, 0, LDA);
    v16h b0 = lds_load_BT(Bt, wn, 0, LDBT);
    v16h b1 = lds_load_BT(Bt, wn + 16, 0, LDBT);
    acc00 = wmma_f32_16x16x32_f16(a0, b0, acc00);
    acc01 = wmma_f32_16x16x32_f16(a0, b1, acc01);
    acc10 = wmma_f32_16x16x32_f16(a1, b0, acc10);
    acc11 = wmma_f32_16x16x32_f16(a1, b1, acc11);
  }

  const int n = lane & 15;
  const int mh = (lane >> 4) << 3;
#pragma unroll
  for (int mi = 0; mi < 2; ++mi) {
#pragma unroll
    for (int r = 0; r < 8; ++r) {
      int m = M0 + wm + mi * 16 + mh + r;
#pragma unroll
      for (int ni = 0; ni < 2; ++ni) {
        const v8f& acc = mi ? (ni ? acc11 : acc10) : (ni ? acc01 : acc00);
        int col = N0 + wn + ni * 16 + n;
        Y[(size_t)m * D_ + col] = acc[r] + bo[col];
      }
    }
  }
}

// ---------------------------------------------------------------------------
extern "C" void kernel_launch(void* const* d_in, const int* in_sizes, int n_in,
                              void* d_out, int out_size, void* d_ws, size_t ws_size,
                              hipStream_t stream) {
  (void)in_sizes; (void)n_in; (void)out_size; (void)ws_size;
  const float* x  = (const float*)d_in[0];
  const float* Wq = (const float*)d_in[1];
  const float* bq = (const float*)d_in[2];
  const float* Wk = (const float*)d_in[3];
  const float* bk = (const float*)d_in[4];
  const float* Wv = (const float*)d_in[5];
  const float* bv = (const float*)d_in[6];
  const float* Wo = (const float*)d_in[7];
  const float* bo = (const float*)d_in[8];
  float* out = (float*)d_out;

  const size_t nqkv = (size_t)B_ * H_ * S_ * HD_;   // 12,582,912 halves
  half_t* qh = (half_t*)d_ws;
  half_t* kh = qh + nqkv;
  half_t* vh = kh + nqkv;
  half_t* oh = vh + nqkv;                           // [B,S,768] f16

  dim3 blk(256);
  dim3 gproj(BS_ / 128, D_ / 64);                   // (128, 12)
  const float qscale = 0.125f;                      // 1/sqrt(64)

  qkv_proj_kernel<<<gproj, blk, 0, stream>>>(x, Wq, bq, qh, qscale);
  qkv_proj_kernel<<<gproj, blk, 0, stream>>>(x, Wk, bk, kh, 1.0f);
  qkv_proj_kernel<<<gproj, blk, 0, stream>>>(x, Wv, bv, vh, 1.0f);

  attn_kernel<<<dim3(B_ * H_, S_ / 128), blk, 0, stream>>>(qh, kh, vh, oh);

  out_proj_kernel<<<gproj, blk, 0, stream>>>(oh, Wo, bo, out);
}
